// MentionScorer_6253472383147
// MI455X (gfx1250) — matrix operand order repaired and checked by reading, not compiled, not run on hardware
//
#include <hip/hip_runtime.h>
#include <hip/hip_bf16.h>
#include <math.h>

typedef __bf16 bf16;
typedef __attribute__((ext_vector_type(16))) __bf16 v16bf;
typedef __attribute__((ext_vector_type(8)))  __bf16 v8bf;
typedef __attribute__((ext_vector_type(8)))  float  v8f;

#define T_TOK 2048
#define D_IN  1024
#define NPAD  160      // hidden 150 padded to 10x16
#define NHID  150
#define NSPAN 20480
#define MAXW  10

// LDS row strides (bf16 elems), padded so lanes r=0..15 hit distinct banks (20-bank step)
#define BROW  40       // 32 data + 8 pad  (80B stride)
#define WROW  168      // 160 data + 8 pad (336B stride)

static __device__ __forceinline__ v16bf cat8(v8bf lo, v8bf hi) {
  return __builtin_shufflevector(lo, hi, 0,1,2,3,4,5,6,7,8,9,10,11,12,13,14,15);
}

// CDNA5 async copy: global -> LDS (16B per lane), tracked by ASYNCcnt.
static __device__ __forceinline__ void async_g2l_b128(const bf16* gptr, bf16* lptr) {
  unsigned lds = (unsigned)(size_t)lptr;   // flat LDS addr carries DS offset in [31:0]
  asm volatile("global_load_async_to_lds_b128 %0, %1, off"
               :: "v"(lds), "v"(gptr) : "memory");
}
static __device__ __forceinline__ void wait_async0() {
#if __has_builtin(__builtin_amdgcn_s_wait_asynccnt)
  __builtin_amdgcn_s_wait_asynccnt(0);
#else
  asm volatile("s_wait_asynccnt 0x0" ::: "memory");
#endif
}

// Force a 2-deep DS-read / WMMA software pipeline for a 10-WMMA chain
// (mask 0x100 = DS read, 0x008 = MFMA/WMMA).
static __device__ __forceinline__ void pipeline_10() {
#if __has_builtin(__builtin_amdgcn_sched_group_barrier)
  __builtin_amdgcn_sched_group_barrier(0x100, 4, 0);     // 4 DS reads ahead
#pragma unroll
  for (int i = 0; i < 9; ++i) {
    __builtin_amdgcn_sched_group_barrier(0x008, 1, 0);   // 1 WMMA
    __builtin_amdgcn_sched_group_barrier(0x100, 2, 0);   // 2 DS reads
  }
  __builtin_amdgcn_sched_group_barrier(0x008, 1, 0);     // last WMMA
#endif
}

// ---------------- prep kernels (VALU, tiny) ----------------

__global__ void cvt_bf16(const float* __restrict__ src, bf16* __restrict__ dst, int n) {
  int i = blockIdx.x * blockDim.x + threadIdx.x;
  int stride = gridDim.x * blockDim.x;
  for (; i < n; i += stride) dst[i] = (bf16)src[i];
}

// out[n*Kpad + k] = (n<150 && k<K) ? W[(row0+k)*150 + n] : 0  for n in [0,160)
__global__ void transpose_pad(const float* __restrict__ W, int row0, int K, int Kpad,
                              bf16* __restrict__ out) {
  int total = NPAD * Kpad;
  int i = blockIdx.x * blockDim.x + threadIdx.x;
  int stride = gridDim.x * blockDim.x;
  for (; i < total; i += stride) {
    int n = i / Kpad, k = i % Kpad;
    float v = (n < NHID && k < K) ? W[(size_t)(row0 + k) * NHID + n] : 0.f;
    out[i] = (bf16)v;
  }
}

// WB[b][j] = ffnn_b1[j] + sum_d width_emb[b][d] * ffnn_W1[3072+d][j]   (0 in pad cols)
__global__ void build_wb(const float* __restrict__ width_emb, const float* __restrict__ fW1,
                         const float* __restrict__ b1, float* __restrict__ wb) {
  int i = blockIdx.x * blockDim.x + threadIdx.x;
  if (i >= 10 * NPAD) return;
  int b = i / NPAD, j = i % NPAD;
  float acc = 0.f;
  if (j < NHID) {
    acc = b1[j];
#pragma unroll
    for (int d = 0; d < 20; ++d)
      acc += width_emb[b * 20 + d] * fW1[(size_t)(3072 + d) * NHID + j];
  }
  wb[i] = acc;
}

// ---------------- K1: token-space GEMMs, M=2048 K=1024 N=160 (WMMA bf16) ----------------
// mode 0: relu(ctx@attn_W1 + attn_b1) -> AttnH1 (bf16)
// mode 1: ctx@W1a -> Pstart; mode 2: ctx@W1b -> Pend; mode 3: emb@W1c -> Pemb (all bf16)
// B staged in LDS via async copy engine, double-buffered, shared by the 4 waves.
__global__ void __launch_bounds__(128)
tok_gemm(const bf16* __restrict__ Actx, const bf16* __restrict__ Aemb,
         const bf16* __restrict__ AW1T, const bf16* __restrict__ W1aT,
         const bf16* __restrict__ W1bT, const bf16* __restrict__ W1cT,
         const float* __restrict__ attn_b1,
         bf16* __restrict__ AttnH1, bf16* __restrict__ Pst,
         bf16* __restrict__ Pen, bf16* __restrict__ Pem) {
  __shared__ bf16 sB[2][NPAD * BROW];

  const int tid  = threadIdx.x;
  const int wave = tid >> 5;
  const int lane = tid & 31;
  const int r = lane & 15, half = lane >> 4;
  const int mtile = blockIdx.x * 4 + wave;   // 0..127
  const int mode  = blockIdx.y;              // 0..3

  const bf16* A  = (mode == 3) ? Aemb : Actx;
  const bf16* WT = (mode == 0) ? AW1T : (mode == 1) ? W1aT : (mode == 2) ? W1bT : W1cT;

  // staging coordinates: 640 16B-chunks (160 rows x 4) over 128 threads, 5 each
  int cn[5], cp[5];
#pragma unroll
  for (int i = 0; i < 5; ++i) { int c = tid + i * 128; cn[i] = c >> 2; cp[i] = c & 3; }

#pragma unroll
  for (int i = 0; i < 5; ++i)
    async_g2l_b128(WT + (size_t)cn[i] * D_IN + cp[i] * 8,
                   &sB[0][cn[i] * BROW + cp[i] * 8]);
  wait_async0();
  __syncthreads();

  v8f acc[10] = {};
  const bf16* Arow = A + (size_t)(mtile * 16 + r) * D_IN;

  // rotated A fragment (loaded one K-step ahead)
  const bf16* ap0 = Arow + half * 8;
  v16bf a = cat8(*(const v8bf*)ap0, *(const v8bf*)(ap0 + 16));

  for (int ks = 0; ks < 32; ++ks) {
    const int cur = ks & 1, nxt = cur ^ 1;
    if (ks < 31) {                         // async-stage next B k-slice (overlaps compute)
#pragma unroll
      for (int i = 0; i < 5; ++i)
        async_g2l_b128(WT + (size_t)cn[i] * D_IN + (ks + 1) * 32 + cp[i] * 8,
                       &sB[nxt][cn[i] * BROW + cp[i] * 8]);
    }
    v16bf an = a;
    if (ks < 31) {                         // issue next A load before the WMMA chain
      const bf16* ap = Arow + (ks + 1) * 32 + half * 8;
      an = cat8(*(const v8bf*)ap, *(const v8bf*)(ap + 16));
    }
    if (ks + 2 < 32) __builtin_prefetch(Arow + (ks + 2) * 32, 0, 3);

    const bf16* bbase = &sB[cur][r * BROW + half * 8];
#pragma unroll
    for (int nt = 0; nt < 10; ++nt) {
      const bf16* bp = bbase + nt * 16 * BROW;
      v16bf b = cat8(*(const v8bf*)bp, *(const v8bf*)(bp + 16));
      acc[nt] = __builtin_amdgcn_wmma_f32_16x16x32_bf16(false, a, false, b,
                                                        (short)0, acc[nt], false, false);
    }
    pipeline_10();
    a = an;
    if (ks < 31) wait_async0();
    __syncthreads();
  }

#pragma unroll
  for (int nt = 0; nt < 10; ++nt) {
#pragma unroll
    for (int rr = 0; rr < 8; ++rr) {
      int m = mtile * 16 + rr + half * 8;
      int n = nt * 16 + r;
      float v = acc[nt][rr];
      if (mode == 0) {
        float bv = (n < NHID) ? attn_b1[n] : 0.f;
        AttnH1[(size_t)m * NPAD + n] = (bf16)fmaxf(v + bv, 0.f);
      } else {
        bf16* O = (mode == 1) ? Pst : (mode == 2) ? Pen : Pem;
        O[(size_t)m * NPAD + n] = (bf16)v;
      }
    }
  }
}

// ---------------- head: out[m] = relu(H@W2 + b2) @ W3 + b3  (WMMA bf16) ----------------
// Whole W2T (160x160 bf16) resident in LDS, async-staged once per block.
__global__ void __launch_bounds__(128)
head_gemm(const bf16* __restrict__ H, const bf16* __restrict__ W2T,
          const float* __restrict__ b2, const float* __restrict__ W3,
          const float* __restrict__ b3, float* __restrict__ out, int mtiles) {
  __shared__ bf16 sW[NPAD * WROW];

  const int tid  = threadIdx.x;
  const int wave = tid >> 5, lane = tid & 31;
  const int r = lane & 15, half = lane >> 4;

  // async-stage W2T: 3200 16B-chunks (160 rows x 20) over 128 threads, 25 each
#pragma unroll
  for (int i = 0; i < 25; ++i) {
    int c = tid + i * 128;
    int n = c / 20, part = c % 20;
    async_g2l_b128(W2T + (size_t)n * NPAD + part * 8, &sW[n * WROW + part * 8]);
  }
  wait_async0();
  __syncthreads();

  const int mtile = blockIdx.x * 4 + wave;
  if (mtile >= mtiles) return;   // wave-uniform

  const bf16* Arow = H + (size_t)(mtile * 16 + r) * NPAD;
  v16bf afr[5];
#pragma unroll
  for (int ks = 0; ks < 5; ++ks) {
    const bf16* ap = Arow + ks * 32 + half * 8;
    afr[ks] = cat8(*(const v8bf*)ap, *(const v8bf*)(ap + 16));
  }

  float part[8];
#pragma unroll
  for (int i = 0; i < 8; ++i) part[i] = 0.f;

#pragma unroll
  for (int nt = 0; nt < 10; ++nt) {
    v8f acc = {};
#pragma unroll
    for (int ks = 0; ks < 5; ++ks) {
      const bf16* bp = &sW[(nt * 16 + r) * WROW + ks * 32 + half * 8];
      v16bf b = cat8(*(const v8bf*)bp, *(const v8bf*)(bp + 16));
      acc = __builtin_amdgcn_wmma_f32_16x16x32_bf16(false, afr[ks], false, b,
                                                    (short)0, acc, false, false);
    }
#if __has_builtin(__builtin_amdgcn_sched_group_barrier)
    __builtin_amdgcn_sched_group_barrier(0x100, 4, 0);
    __builtin_amdgcn_sched_group_barrier(0x008, 1, 0);
    __builtin_amdgcn_sched_group_barrier(0x100, 2, 0);
    __builtin_amdgcn_sched_group_barrier(0x008, 1, 0);
    __builtin_amdgcn_sched_group_barrier(0x100, 2, 0);
    __builtin_amdgcn_sched_group_barrier(0x008, 1, 0);
    __builtin_amdgcn_sched_group_barrier(0x100, 2, 0);
    __builtin_amdgcn_sched_group_barrier(0x008, 1, 0);
    __builtin_amdgcn_sched_group_barrier(0x008, 1, 0);
#endif
    int n = nt * 16 + r;
    float b2v = (n < NHID) ? b2[n] : 0.f;
    float w3v = (n < NHID) ? W3[n] : 0.f;
#pragma unroll
    for (int rr = 0; rr < 8; ++rr)
      part[rr] += fmaxf(acc[rr] + b2v, 0.f) * w3v;
  }

#pragma unroll
  for (int rr = 0; rr < 8; ++rr) {
    float p = part[rr];
    p += __shfl_xor(p, 1, 32);
    p += __shfl_xor(p, 2, 32);
    p += __shfl_xor(p, 4, 32);
    p += __shfl_xor(p, 8, 32);
    if ((lane & 15) == 0)
      out[mtile * 16 + rr + half * 8] = p + b3[0];
  }
}

// ---------------- K3: per-span softmax + projected gather -> SpanH1 bf16 ----------------
__global__ void __launch_bounds__(256)
span_gather(const int* __restrict__ starts, const int* __restrict__ widths,
            const float* __restrict__ attns, const bf16* __restrict__ Pst,
            const bf16* __restrict__ Pen, const bf16* __restrict__ Pem,
            const float* __restrict__ WB, bf16* __restrict__ SpanH1) {
  const int wave = threadIdx.x >> 5, lane = threadIdx.x & 31;
  const int n = blockIdx.x * 8 + wave;
  if (n >= NSPAN) return;
  const int s0 = starts[n];
  const int wd = widths[n];
  const int len = wd + 1;
  const int en = s0 + wd;

  float a[MAXW];
  float mx = -1e30f;
#pragma unroll
  for (int l = 0; l < MAXW; ++l) {
    float v = (l < len) ? attns[min(s0 + l, T_TOK - 1)] : -1e30f;
    a[l] = v; mx = fmaxf(mx, v);
  }
  float s = 0.f;
#pragma unroll
  for (int l = 0; l < MAXW; ++l) {
    float e = (l < len) ? __expf(a[l] - mx) : 0.f;
    a[l] = e; s += e;
  }
  float inv = 1.f / s;
  int bucket = (len <= 4) ? (len - 1) : ((len <= 7) ? 4 : 5);
  const float* wbrow = WB + bucket * NPAD;

#pragma unroll
  for (int c = 0; c < 5; ++c) {
    int j = c * 32 + lane;
    float acc = (float)Pst[(size_t)s0 * NPAD + j] + (float)Pen[(size_t)en * NPAD + j]
              + wbrow[j];
#pragma unroll
    for (int l = 0; l < MAXW; ++l)
      if (l < len) acc += (a[l] * inv) * (float)Pem[(size_t)(s0 + l) * NPAD + j];
    SpanH1[(size_t)n * NPAD + j] = (bf16)fmaxf(acc, 0.f);
  }
}

// ---------------- launch ----------------

extern "C" void kernel_launch(void* const* d_in, const int* in_sizes, int n_in,
                              void* d_out, int out_size, void* d_ws, size_t ws_size,
                              hipStream_t stream) {
  const float* ctx  = (const float*)d_in[0];
  const float* emb  = (const float*)d_in[1];
  const int*   sst  = (const int*)d_in[2];
  const int*   swd  = (const int*)d_in[3];
  const float* aW1  = (const float*)d_in[4];
  const float* ab1  = (const float*)d_in[5];
  const float* aW2  = (const float*)d_in[6];
  const float* ab2  = (const float*)d_in[7];
  const float* aW3  = (const float*)d_in[8];
  const float* ab3  = (const float*)d_in[9];
  const float* wemb = (const float*)d_in[10];
  const float* fW1  = (const float*)d_in[11];
  const float* fb1  = (const float*)d_in[12];
  const float* fW2  = (const float*)d_in[13];
  const float* fb2  = (const float*)d_in[14];
  const float* fW3  = (const float*)d_in[15];
  const float* fb3  = (const float*)d_in[16];

  char* base = (char*)d_ws;
  size_t off = 0;
  auto alloc = [&](size_t bytes) -> void* {
    void* p = base + off;
    off = (off + bytes + 255) & ~(size_t)255;
    return p;
  };

  bf16*  ctx_bf = (bf16*)alloc((size_t)T_TOK * D_IN * 2);
  bf16*  emb_bf = (bf16*)alloc((size_t)T_TOK * D_IN * 2);
  bf16*  AW1T   = (bf16*)alloc((size_t)NPAD * D_IN * 2);
  bf16*  W1aT   = (bf16*)alloc((size_t)NPAD * D_IN * 2);
  bf16*  W1bT   = (bf16*)alloc((size_t)NPAD * D_IN * 2);
  bf16*  W1cT   = (bf16*)alloc((size_t)NPAD * D_IN * 2);
  bf16*  AW2T   = (bf16*)alloc((size_t)NPAD * NPAD * 2);
  bf16*  FW2T   = (bf16*)alloc((size_t)NPAD * NPAD * 2);
  float* WB     = (float*)alloc((size_t)10 * NPAD * 4);
  bf16*  AttnH1 = (bf16*)alloc((size_t)T_TOK * NPAD * 2);
  bf16*  Pst    = (bf16*)alloc((size_t)T_TOK * NPAD * 2);
  bf16*  Pen    = (bf16*)alloc((size_t)T_TOK * NPAD * 2);
  bf16*  Pem    = (bf16*)alloc((size_t)T_TOK * NPAD * 2);
  float* attns  = (float*)alloc((size_t)T_TOK * 4);
  bf16*  SpanH1 = (bf16*)alloc((size_t)NSPAN * NPAD * 2);

  // 1) fp32 -> bf16 activations
  cvt_bf16<<<1024, 256, 0, stream>>>(ctx, ctx_bf, T_TOK * D_IN);
  cvt_bf16<<<1024, 256, 0, stream>>>(emb, emb_bf, T_TOK * D_IN);

  // 2) transposed, padded bf16 weights (B matrices laid out N-major, K contiguous)
  transpose_pad<<<640, 256, 0, stream>>>(aW1, 0,    D_IN, D_IN, AW1T);
  transpose_pad<<<640, 256, 0, stream>>>(fW1, 0,    D_IN, D_IN, W1aT);
  transpose_pad<<<640, 256, 0, stream>>>(fW1, 1024, D_IN, D_IN, W1bT);
  transpose_pad<<<640, 256, 0, stream>>>(fW1, 2048, D_IN, D_IN, W1cT);
  transpose_pad<<<100, 256, 0, stream>>>(aW2, 0,    NHID, NPAD, AW2T);
  transpose_pad<<<100, 256, 0, stream>>>(fW2, 0,    NHID, NPAD, FW2T);

  // 3) width-bucket rows projected through W1 tail (+ b1 folded in)
  build_wb<<<7, 256, 0, stream>>>(wemb, fW1, fb1, WB);

  // 4) token GEMMs: AttnH1, Pstart, Pend, Pemb (WMMA bf16, async-LDS-staged B)
  tok_gemm<<<dim3(32, 4), 128, 0, stream>>>(ctx_bf, emb_bf, AW1T, W1aT, W1bT, W1cT,
                                            ab1, AttnH1, Pst, Pen, Pem);

  // 5) attn FFNN layers 2-3 -> per-token logits
  head_gemm<<<32, 128, 0, stream>>>(AttnH1, AW2T, ab2, aW3, ab3, attns, T_TOK / 16);

  // 6) per-span softmax + gather in projected space -> SpanH1 (= relu of FFNN layer 1)
  span_gather<<<NSPAN / 8, 256, 0, stream>>>(sst, swd, attns, Pst, Pen, Pem, WB, SpanH1);

  // 7) mention FFNN layers 2-3 -> scores
  head_gemm<<<NSPAN / 64, 128, 0, stream>>>(SpanH1, FW2T, fb2, fW3, fb3,
                                            (float*)d_out, NSPAN / 16);
}
